// ValidationNet_72868415144152
// MI455X (gfx1250) — compile-verified, matrix operand order checked by reading, not run on hardware
//
#include <hip/hip_runtime.h>
#include <hip/hip_bf16.h>

typedef __attribute__((ext_vector_type(16))) _Float16 v16h;
typedef __attribute__((ext_vector_type(8)))  float    v8f;

union F16Frag {       // 16 f16 = 32 bytes = two 16-byte chunks
    v16h v;
    uint4 u[2];
};

#define QTOT   4096
#define INDIM  512
#define DDIM   128
#define NTOT   32768
#define NCH    32
#define CHUNK  (NTOT / NCH)      // 1024 columns per wave

#define LOG2E  1.4426950408889634f
#define LN2    0.6931471805599453f
// ln(N) + D*(ln(bw) + 0.5*ln(2*pi)),  N=32768, bw=0.5, D=128
#define OUTCONST 39.29850084692429f

// single-exp2 online logsumexp merge of (m,s) with one new sample a2
__device__ __forceinline__ void lse_update(float& m, float& s, float a2) {
    float d  = a2 - m;
    float e  = __builtin_amdgcn_exp2f(-__builtin_fabsf(d));
    bool  up = d > 0.0f;
    s = up ? __builtin_fmaf(s, e, 1.0f) : (s + e);
    m = up ? a2 : m;
}

// single-exp2 merge of two (m,s) partials
__device__ __forceinline__ void lse_merge(float& m, float& s, float mo, float so) {
    float d  = mo - m;
    float e  = __builtin_amdgcn_exp2f(-__builtin_fabsf(d));
    bool  up = d > 0.0f;
    s = up ? __builtin_fmaf(s, e, so) : __builtin_fmaf(so, e, s);
    m = up ? mo : m;
}

// ---------------- Kernel 1: latent = x @ W^T + b ; q2 = ||latent||^2 -------
__global__ __launch_bounds__(128) void latent_kernel(
    const float* __restrict__ x, const float* __restrict__ W,
    const float* __restrict__ b, _Float16* __restrict__ latf16,
    float* __restrict__ q2)
{
    __shared__ float xs[INDIM];
    __shared__ float red[DDIM];
    const int q = blockIdx.x;
    const int d = threadIdx.x;

    #pragma unroll
    for (int i = 0; i < INDIM / 128; ++i)
        xs[d + i * 128] = x[q * INDIM + d + i * 128];
    __syncthreads();

    const float4* wr = (const float4*)(W + d * INDIM);
    float acc = b[d];
    #pragma unroll 4
    for (int k = 0; k < INDIM / 4; ++k) {
        float4 w4 = wr[k];
        acc = fmaf(xs[4 * k + 0], w4.x, acc);
        acc = fmaf(xs[4 * k + 1], w4.y, acc);
        acc = fmaf(xs[4 * k + 2], w4.z, acc);
        acc = fmaf(xs[4 * k + 3], w4.w, acc);
    }
    latf16[q * DDIM + d] = (_Float16)acc;

    red[d] = acc * acc;
    __syncthreads();
    for (int s = 64; s > 0; s >>= 1) {
        if (d < s) red[d] += red[d + s];
        __syncthreads();
    }
    if (d == 0) q2[q] = red[0];
}

// ---------------- Kernel 2: points -> f16 ; p2 = ||points||^2 --------------
__global__ __launch_bounds__(128) void points_kernel(
    const float* __restrict__ pts, _Float16* __restrict__ pf16,
    float* __restrict__ p2)
{
    __shared__ float red[DDIM];
    const int n = blockIdx.x;
    const int d = threadIdx.x;
    float v = pts[n * DDIM + d];
    pf16[n * DDIM + d] = (_Float16)v;
    red[d] = v * v;
    __syncthreads();
    for (int s = 64; s > 0; s >>= 1) {
        if (d < s) red[d] += red[d + s];
        __syncthreads();
    }
    if (d == 0) p2[n] = red[0];
}

// load one 16-col tile's B fragments and issue 4 independent WMMAs
__device__ __forceinline__ void tile_mma(
    const _Float16* __restrict__ pf16, const float* __restrict__ p2,
    int n, int kb0, const F16Frag A[4], v8f c[4], float& pterm)
{
    const _Float16* prow = pf16 + (size_t)n * DDIM;
    pterm = (2.0f * LOG2E) * p2[n];

    F16Frag B0, B1, B2, B3;
    B0.u[0] = *(const uint4*)(prow + kb0);
    B0.u[1] = *(const uint4*)(prow + kb0 + 16);
    B1.u[0] = *(const uint4*)(prow + 32 + kb0);
    B1.u[1] = *(const uint4*)(prow + 32 + kb0 + 16);
    B2.u[0] = *(const uint4*)(prow + 64 + kb0);
    B2.u[1] = *(const uint4*)(prow + 64 + kb0 + 16);
    B3.u[0] = *(const uint4*)(prow + 96 + kb0);
    B3.u[1] = *(const uint4*)(prow + 96 + kb0 + 16);
    __builtin_prefetch(prow + 16 * DDIM, 0, 3);   // next tile's row

    v8f z = {};
    c[0] = __builtin_amdgcn_wmma_f32_16x16x32_f16(
               false, A[0].v, false, B0.v, (short)0, z, false, false);
    c[1] = __builtin_amdgcn_wmma_f32_16x16x32_f16(
               false, A[1].v, false, B1.v, (short)0, z, false, false);
    c[2] = __builtin_amdgcn_wmma_f32_16x16x32_f16(
               false, A[2].v, false, B2.v, (short)0, z, false, false);
    c[3] = __builtin_amdgcn_wmma_f32_16x16x32_f16(
               false, A[3].v, false, B3.v, (short)0, z, false, false);
}

// consume a tile's WMMA results: online base-2 logsumexp update per row
__device__ __forceinline__ void tile_epilogue(
    const v8f c[4], float pterm, const float qterm[8], float m[8], float s[8])
{
    #pragma unroll
    for (int r = 0; r < 8; ++r) {
        float S  = (c[0][r] + c[1][r]) + (c[2][r] + c[3][r]);
        float a2 = fmaf(4.0f * LOG2E, S, -(qterm[r] + pterm));
        lse_update(m[r], s[r], a2);
    }
}

// ---------------- Kernel 3: fused WMMA GEMM + online logsumexp -------------
// One wave (32 threads) per (16-query tile, 1024-column chunk).
// Software-pipelined: tile t's epilogue runs after tile t+1's WMMAs issue,
// so WMMA->VALU hazard slots are filled with useful work.
__global__ __launch_bounds__(32) void kde_main(
    const _Float16* __restrict__ latf16, const float* __restrict__ q2,
    const _Float16* __restrict__ pf16,   const float* __restrict__ p2,
    float* __restrict__ pm, float* __restrict__ ps)
{
    const int qt   = blockIdx.x;          // 0..255
    const int ch   = blockIdx.y;          // 0..NCH-1
    const int q0   = qt * 16;
    const int n0   = ch * CHUNK;
    const int lane = threadIdx.x;
    const int half = lane >> 4;           // K-half selector (wave32 layout)
    const int lm   = lane & 15;           // row (A) / column (B,C) index
    const int kb0  = 8 * half;

    // A fragments: 16x128 f16 latent tile, K split in 4 fragments of 32.
    F16Frag A[4];
    {
        const _Float16* arow = latf16 + (size_t)(q0 + lm) * DDIM;
        #pragma unroll
        for (int f = 0; f < 4; ++f) {
            const int kb = 32 * f + 8 * half;
            A[f].u[0] = *(const uint4*)(arow + kb);
            A[f].u[1] = *(const uint4*)(arow + kb + 16);
        }
    }

    // C/D layout: VGPR r -> row q0 + r + 8*half, column tile + lm
    float qterm[8];
    #pragma unroll
    for (int r = 0; r < 8; ++r)
        qterm[r] = (2.0f * LOG2E) * q2[q0 + r + 8 * half];

    float m[8], s[8];
    #pragma unroll
    for (int r = 0; r < 8; ++r) { m[r] = -__builtin_inff(); s[r] = 0.0f; }

    v8f ca[4], cb[4];
    float ptA, ptB;

    // pipeline prologue: issue tiles 0 and 16, then drain tile 0
    tile_mma(pf16, p2, n0 + lm,      kb0, A, ca, ptA);
    tile_mma(pf16, p2, n0 + 16 + lm, kb0, A, cb, ptB);
    tile_epilogue(ca, ptA, qterm, m, s);

    // steady state: 2 tiles per iteration (CHUNK/16 = 64 tiles total)
    for (int t = 32; t < CHUNK; t += 32) {
        tile_mma(pf16, p2, n0 + t + lm, kb0, A, ca, ptA);
        tile_epilogue(cb, ptB, qterm, m, s);
        tile_mma(pf16, p2, n0 + t + 16 + lm, kb0, A, cb, ptB);
        tile_epilogue(ca, ptA, qterm, m, s);
    }
    // drain last pending tile
    tile_epilogue(cb, ptB, qterm, m, s);

    // merge the 16 per-lane partials of each row (xor masks < 16 keep halves)
    #pragma unroll
    for (int r = 0; r < 8; ++r) {
        #pragma unroll
        for (int mask = 1; mask < 16; mask <<= 1) {
            float mo = __shfl_xor(m[r], mask, 32);
            float so = __shfl_xor(s[r], mask, 32);
            lse_merge(m[r], s[r], mo, so);
        }
    }

    if (lm == 0) {   // lane 0 writes rows 0..7, lane 16 writes rows 8..15
        #pragma unroll
        for (int r = 0; r < 8; ++r) {
            const int row = q0 + r + 8 * half;
            pm[(size_t)row * NCH + ch] = m[r];
            ps[(size_t)row * NCH + ch] = s[r];
        }
    }
}

// ---------------- Kernel 4: merge chunk partials, finish -------------------
__global__ __launch_bounds__(256) void finalize_kernel(
    const float* __restrict__ pm, const float* __restrict__ ps,
    float* __restrict__ out)
{
    const int q = blockIdx.x * blockDim.x + threadIdx.x;
    if (q >= QTOT) return;
    float M = pm[(size_t)q * NCH];
    float S = ps[(size_t)q * NCH];
    #pragma unroll 4
    for (int c = 1; c < NCH; ++c) {
        float mo = pm[(size_t)q * NCH + c];
        float so = ps[(size_t)q * NCH + c];
        lse_merge(M, S, mo, so);
    }
    const float lse = (M + __log2f(S)) * LN2;   // natural-log logsumexp
    out[q] = OUTCONST - lse;                     // = -logkde
}

// ---------------- Launch ---------------------------------------------------
extern "C" void kernel_launch(void* const* d_in, const int* in_sizes, int n_in,
                              void* d_out, int out_size, void* d_ws, size_t ws_size,
                              hipStream_t stream) {
    const float* x      = (const float*)d_in[0];
    const float* W      = (const float*)d_in[1];
    const float* b      = (const float*)d_in[2];
    const float* points = (const float*)d_in[3];
    float* out = (float*)d_out;

    // workspace carve (all offsets 256B-aligned)
    char* w = (char*)d_ws;
    _Float16* latf16 = (_Float16*)w;  w += (size_t)QTOT * DDIM * sizeof(_Float16); // 1 MB
    _Float16* pf16   = (_Float16*)w;  w += (size_t)NTOT * DDIM * sizeof(_Float16); // 8 MB
    float* q2 = (float*)w;            w += (size_t)QTOT * sizeof(float);
    float* p2 = (float*)w;            w += (size_t)NTOT * sizeof(float);
    float* pm = (float*)w;            w += (size_t)QTOT * NCH * sizeof(float);
    float* ps = (float*)w;            w += (size_t)QTOT * NCH * sizeof(float);

    latent_kernel<<<QTOT, 128, 0, stream>>>(x, W, b, latf16, q2);
    points_kernel<<<NTOT, 128, 0, stream>>>(points, pf16, p2);

    dim3 grid(QTOT / 16, NCH);
    kde_main<<<grid, 32, 0, stream>>>(latf16, q2, pf16, p2, pm, ps);

    finalize_kernel<<<(QTOT + 255) / 256, 256, 0, stream>>>(pm, ps, out);
}
